// SarvamMoESparseMoeBlock_73847667687614
// MI455X (gfx1250) — compile-verified, matrix operand order checked by reading, not run on hardware
//
#include <hip/hip_runtime.h>
#include <hip/hip_bf16.h>

// ---------------- problem constants (from reference) ----------------
#define T_TOK   2048      // B*S tokens
#define H_DIM   2048
#define I_DIM   1024
#define NEXP    32
#define TOPK    6
#define NGRP    8
#define EPG     (NEXP / NGRP)   // experts per group = 4
#define TOPG    4
#define RSCALE  2.5f
#define M_TILE  256

typedef __attribute__((ext_vector_type(16))) __bf16 bf16x16;
typedef __attribute__((ext_vector_type(8)))  float  f32x8;

// ---------------- workspace layout (bytes) ----------------
// [0,128)                 counts[32]
// [128, 128+34*4)         offs[34] (prefix sums incl. shared expert @32)
// [512, 512+256K)         tok_list[32][2048]
// [512+256K, 512+512K)    wgt_list[32][2048]
// [1M, 1M + ~15360*1024*2) z scratch, bf16, compacted rows per expert
#define WS_TOK_OFF   512
#define WS_WGT_OFF   (512 + NEXP * T_TOK * 4)
#define WS_Z_OFF     (1u << 20)

// =====================================================================
// Kernel 0: zero expert counts (y is fully overwritten by shared-down)
// =====================================================================
__global__ void init_counts_kernel(int* __restrict__ counts) {
    if (threadIdx.x < NEXP) counts[threadIdx.x] = 0;
}

// =====================================================================
// Kernel 1: gate logits + sigmoid + group-limited top-k routing.
// One wave per token; lane e computes logits[t][e].
// =====================================================================
__global__ __launch_bounds__(256) void gate_route_kernel(
    const float* __restrict__ x, const float* __restrict__ gate_w,
    const float* __restrict__ bias, float* __restrict__ out_logits,
    int* __restrict__ out_idx, int* __restrict__ counts,
    int* __restrict__ tok_list, float* __restrict__ wgt_list)
{
    __shared__ float scS[8][NEXP];   // sigmoid scores
    __shared__ float srS[8][NEXP];   // biased routing scores
    const int w    = threadIdx.x >> 5;
    const int lane = threadIdx.x & 31;
    const int t    = blockIdx.x * 8 + w;

    const float4* xr = (const float4*)(x + (size_t)t * H_DIM);
    const float4* gr = (const float4*)(gate_w + (size_t)lane * H_DIM);
    float acc = 0.f;
    #pragma unroll 4
    for (int i = 0; i < H_DIM / 4; ++i) {
        float4 a = xr[i]; float4 b = gr[i];
        acc += a.x * b.x + a.y * b.y + a.z * b.z + a.w * b.w;
    }
    out_logits[(size_t)t * NEXP + lane] = acc;
    float sc = 1.f / (1.f + __expf(-acc));
    scS[w][lane] = sc;
    srS[w][lane] = sc + bias[lane];
    __syncthreads();

    if (lane == 0) {
        const float* sr  = srS[w];
        const float* scp = scS[w];
        // group score = sum of top-2 experts in group
        float gs[NGRP];
        #pragma unroll
        for (int g = 0; g < NGRP; ++g) {
            float m1 = -1e30f, m2 = -1e30f;
            #pragma unroll
            for (int j = 0; j < EPG; ++j) {
                float v = sr[g * EPG + j];
                if (v > m1) { m2 = m1; m1 = v; } else if (v > m2) { m2 = v; }
            }
            gs[g] = m1 + m2;
        }
        // top TOPG groups (strict > keeps first index on ties, like jax top_k)
        bool gsel[NGRP];
        #pragma unroll
        for (int g = 0; g < NGRP; ++g) gsel[g] = false;
        for (int it = 0; it < TOPG; ++it) {
            float best = -1e30f; int bi = 0;
            #pragma unroll
            for (int g = 0; g < NGRP; ++g)
                if (!gsel[g] && gs[g] > best) { best = gs[g]; bi = g; }
            gsel[bi] = true;
        }
        // top TOPK experts among eligible groups, by biased score
        bool esel[NEXP];
        #pragma unroll
        for (int e = 0; e < NEXP; ++e) esel[e] = false;
        int   idx[TOPK];
        float tw[TOPK];
        float sum = 0.f;
        for (int k = 0; k < TOPK; ++k) {
            float best = -1e30f; int bi = 0;
            #pragma unroll
            for (int e = 0; e < NEXP; ++e)
                if (gsel[e >> 2] && !esel[e] && sr[e] > best) { best = sr[e]; bi = e; }
            esel[bi] = true;
            idx[k] = bi;
            tw[k]  = scp[bi];          // weights from UN-biased sigmoid scores
            sum   += tw[k];
        }
        float inv = RSCALE / (sum + 1e-20f);
        for (int k = 0; k < TOPK; ++k) {
            out_idx[(size_t)t * TOPK + k] = idx[k];
            int pos = atomicAdd(&counts[idx[k]], 1);
            tok_list[idx[k] * T_TOK + pos] = t;
            wgt_list[idx[k] * T_TOK + pos] = tw[k] * inv;
        }
    }
}

// =====================================================================
// Kernel 2: prefix sums of per-expert counts (33 entries incl. shared)
// =====================================================================
__global__ void prefix_kernel(const int* __restrict__ counts, int* __restrict__ offs) {
    if (threadIdx.x == 0 && blockIdx.x == 0) {
        int acc = 0;
        for (int e = 0; e <= NEXP; ++e) {
            offs[e] = acc;
            acc += (e < NEXP) ? counts[e] : T_TOK;
        }
        offs[NEXP + 1] = acc;
    }
}

// =====================================================================
// WMMA tile loaders (CDNA5 16x16x32 bf16 layouts)
// A 16x32 bf16: lanes 0-15 = rows M, K in {0..7,16..23}; lanes 16-31 K+8
// B 32x16 bf16: columns in lanes, same per-lane K ordering
// =====================================================================
__device__ __forceinline__ bf16x16 load_a_lds(const __bf16* __restrict__ rowp, int kh) {
    bf16x16 v;
    #pragma unroll
    for (int j = 0; j < 8; ++j) v[j]     = rowp[kh + j];
    #pragma unroll
    for (int j = 0; j < 8; ++j) v[8 + j] = rowp[16 + kh + j];
    return v;
}

__device__ __forceinline__ bf16x16 load_b_f32(const float* __restrict__ b0, int ldb, int kh) {
    bf16x16 v;
    #pragma unroll
    for (int j = 0; j < 8; ++j) v[j]     = (__bf16)b0[(size_t)(kh + j) * ldb];
    #pragma unroll
    for (int j = 0; j < 8; ++j) v[8 + j] = (__bf16)b0[(size_t)(16 + kh + j) * ldb];
    return v;
}

// =====================================================================
// Kernel 3: gate/up GEMMs + SiLU -> z (bf16, compacted rows per expert)
// grid = (I/128, T/256, 33). 512 threads = 16 waves = 8 (M-pair) x 2 (N-half).
// Each wave: 2 M-stripes (16 rows each) x 4 N-tiles; B tiles are reused
// across both stripes. A panel is register-prefetched one k-chunk ahead.
// =====================================================================
__global__ __launch_bounds__(512) void expert_gateup_kernel(
    const float* __restrict__ x, const float* __restrict__ Wg,
    const float* __restrict__ Wu, const float* __restrict__ Wg_s,
    const float* __restrict__ Wu_s, const int* __restrict__ counts,
    const int* __restrict__ offs, const int* __restrict__ tok_list,
    __bf16* __restrict__ zws)
{
    const int  e       = blockIdx.z;
    const bool sharedE = (e == NEXP);
    const int  cnt     = sharedE ? T_TOK : counts[e];
    const int  m0      = blockIdx.y * M_TILE;
    if (m0 >= cnt) return;

    __shared__ __bf16 As[M_TILE * 36];   // padded stride avoids bank conflicts
    __shared__ int    tokS[M_TILE];

    const int tid  = threadIdx.x;
    const int lane = tid & 31;
    const int wid  = tid >> 5;
    const int kh   = (lane >> 4) << 3;   // 0 or 8
    const int colL = lane & 15;
    const int mbase  = (wid >> 1) * 32;  // first of the wave's two stripes
    const int nhalf  = (wid & 1) * 4;
    const int nblk   = blockIdx.x * 128;

    if (tid < M_TILE) {
        int mg = m0 + tid;
        tokS[tid] = sharedE ? ((mg < T_TOK) ? mg : 0)
                            : ((mg < cnt) ? tok_list[e * T_TOK + mg] : 0);
    }
    __syncthreads();

    // staging: this thread's column is its lane; its rows are wid + 16*i
    unsigned ofs[16];
    #pragma unroll
    for (int i = 0; i < 16; ++i)
        ofs[i] = (unsigned)tokS[wid + 16 * i] * H_DIM + lane;
    float xa[16];
    #pragma unroll
    for (int i = 0; i < 16; ++i) xa[i] = x[ofs[i]];      // prefetch chunk 0

    const float* bg_base = sharedE ? Wg_s : (Wg + (size_t)e * H_DIM * I_DIM);
    const float* bu_base = sharedE ? Wu_s : (Wu + (size_t)e * H_DIM * I_DIM);

    f32x8 accg[2][4], accu[2][4];
    #pragma unroll
    for (int s = 0; s < 2; ++s)
        #pragma unroll
        for (int nt = 0; nt < 4; ++nt)
            #pragma unroll
            for (int j = 0; j < 8; ++j) { accg[s][nt][j] = 0.f; accu[s][nt][j] = 0.f; }

    for (int kc = 0; kc < H_DIM; kc += 32) {
        __syncthreads();                     // readers of previous panel done
        #pragma unroll
        for (int i = 0; i < 16; ++i)
            As[(wid + 16 * i) * 36 + lane] = (__bf16)xa[i];
        __syncthreads();
        if (kc + 32 < H_DIM) {               // issue next chunk's loads now
            #pragma unroll
            for (int i = 0; i < 16; ++i) xa[i] = x[ofs[i] + kc + 32];
        }

        bf16x16 a0 = load_a_lds(As + (mbase + colL) * 36, kh);
        bf16x16 a1 = load_a_lds(As + (mbase + 16 + colL) * 36, kh);
        #pragma unroll
        for (int nt = 0; nt < 4; ++nt) {
            int ncol = nblk + (nhalf + nt) * 16 + colL;
            bf16x16 vbg = load_b_f32(bg_base + (size_t)kc * I_DIM + ncol, I_DIM, kh);
            bf16x16 vbu = load_b_f32(bu_base + (size_t)kc * I_DIM + ncol, I_DIM, kh);
            accg[0][nt] = __builtin_amdgcn_wmma_f32_16x16x32_bf16(
                false, a0, false, vbg, (short)0, accg[0][nt], false, false);
            accg[1][nt] = __builtin_amdgcn_wmma_f32_16x16x32_bf16(
                false, a1, false, vbg, (short)0, accg[1][nt], false, false);
            accu[0][nt] = __builtin_amdgcn_wmma_f32_16x16x32_bf16(
                false, a0, false, vbu, (short)0, accu[0][nt], false, false);
            accu[1][nt] = __builtin_amdgcn_wmma_f32_16x16x32_bf16(
                false, a1, false, vbu, (short)0, accu[1][nt], false, false);
        }
    }

    const int zbase = offs[e];
    #pragma unroll
    for (int s = 0; s < 2; ++s) {
        #pragma unroll
        for (int nt = 0; nt < 4; ++nt) {
            #pragma unroll
            for (int j = 0; j < 8; ++j) {
                int m  = mbase + s * 16 + ((lane < 16) ? j : j + 8);
                int mg = m0 + m;
                if (mg < cnt) {
                    float gv = accg[s][nt][j], uv = accu[s][nt][j];
                    float zv = (gv / (1.f + __expf(-gv))) * uv;   // silu(g)*u
                    zws[(size_t)(zbase + mg) * I_DIM + nblk + (nhalf + nt) * 16 + colL]
                        = (__bf16)zv;
                }
            }
        }
    }
}

// =====================================================================
// Kernel 4: down GEMM.
//  SHARED=true : shared expert, every row valid, PLAIN stores (defines y)
//  SHARED=false: routed experts, scale by routing weight, atomic adds
// grid = (H/128, T/256, [NEXP]).
// =====================================================================
template <bool SHARED>
__global__ __launch_bounds__(512) void down_kernel(
    const float* __restrict__ Wd_base,   // routed: Wd; shared: Wd_s
    const int* __restrict__ counts, const int* __restrict__ offs,
    const int* __restrict__ tok_list, const float* __restrict__ wgt_list,
    const __bf16* __restrict__ zws, float* __restrict__ y)
{
    const int e   = SHARED ? NEXP : blockIdx.z;
    const int cnt = SHARED ? T_TOK : counts[e];
    const int m0  = blockIdx.y * M_TILE;
    if (m0 >= cnt) return;

    __shared__ __bf16 Zs[M_TILE * 36];
    __shared__ int    tokS[M_TILE];
    __shared__ float  wgtS[M_TILE];

    const int tid  = threadIdx.x;
    const int lane = tid & 31;
    const int wid  = tid >> 5;
    const int kh   = (lane >> 4) << 3;
    const int colL = lane & 15;
    const int mbase = (wid >> 1) * 32;
    const int nhalf = (wid & 1) * 4;
    const int nblk  = blockIdx.x * 128;

    if (!SHARED && tid < M_TILE) {
        int mg = m0 + tid;
        if (mg < cnt) { tokS[tid] = tok_list[e * T_TOK + mg];
                        wgtS[tid] = wgt_list[e * T_TOK + mg]; }
        else          { tokS[tid] = 0; wgtS[tid] = 0.f; }
    }
    __syncthreads();

    const int zbase = offs[e];

    unsigned zofs[16];
    #pragma unroll
    for (int i = 0; i < 16; ++i)
        zofs[i] = (unsigned)(zbase + m0 + wid + 16 * i) * I_DIM + lane;
    __bf16 za[16];
    #pragma unroll
    for (int i = 0; i < 16; ++i) za[i] = zws[zofs[i]];

    const float* bd = SHARED ? Wd_base : (Wd_base + (size_t)e * I_DIM * H_DIM);

    f32x8 acc[2][4];
    #pragma unroll
    for (int s = 0; s < 2; ++s)
        #pragma unroll
        for (int nt = 0; nt < 4; ++nt)
            #pragma unroll
            for (int j = 0; j < 8; ++j) acc[s][nt][j] = 0.f;

    for (int kc = 0; kc < I_DIM; kc += 32) {
        __syncthreads();
        #pragma unroll
        for (int i = 0; i < 16; ++i)
            Zs[(wid + 16 * i) * 36 + lane] = za[i];
        __syncthreads();
        if (kc + 32 < I_DIM) {
            #pragma unroll
            for (int i = 0; i < 16; ++i) za[i] = zws[zofs[i] + kc + 32];
        }

        bf16x16 a0 = load_a_lds(Zs + (mbase + colL) * 36, kh);
        bf16x16 a1 = load_a_lds(Zs + (mbase + 16 + colL) * 36, kh);
        #pragma unroll
        for (int nt = 0; nt < 4; ++nt) {
            int ncol = nblk + (nhalf + nt) * 16 + colL;
            bf16x16 vb = load_b_f32(bd + (size_t)kc * H_DIM + ncol, H_DIM, kh);
            acc[0][nt] = __builtin_amdgcn_wmma_f32_16x16x32_bf16(
                false, a0, false, vb, (short)0, acc[0][nt], false, false);
            acc[1][nt] = __builtin_amdgcn_wmma_f32_16x16x32_bf16(
                false, a1, false, vb, (short)0, acc[1][nt], false, false);
        }
    }

    #pragma unroll
    for (int s = 0; s < 2; ++s) {
        #pragma unroll
        for (int nt = 0; nt < 4; ++nt) {
            #pragma unroll
            for (int j = 0; j < 8; ++j) {
                int m  = mbase + s * 16 + ((lane < 16) ? j : j + 8);
                int mg = m0 + m;
                int col = nblk + (nhalf + nt) * 16 + colL;
                if (SHARED) {
                    // defines y (runs before routed pass) - no zero pass needed
                    y[(size_t)mg * H_DIM + col] = acc[s][nt][j];
                } else if (mg < cnt) {
                    atomicAdd(&y[(size_t)tokS[m] * H_DIM + col],
                              acc[s][nt][j] * wgtS[m]);
                }
            }
        }
    }
}

// =====================================================================
// Launcher
// =====================================================================
extern "C" void kernel_launch(void* const* d_in, const int* in_sizes, int n_in,
                              void* d_out, int out_size, void* d_ws, size_t ws_size,
                              hipStream_t stream) {
    const float* x      = (const float*)d_in[0];
    const float* gate_w = (const float*)d_in[1];
    const float* bias   = (const float*)d_in[2];
    const float* Wg     = (const float*)d_in[3];
    const float* Wu     = (const float*)d_in[4];
    const float* Wd     = (const float*)d_in[5];
    const float* Wg_s   = (const float*)d_in[6];
    const float* Wu_s   = (const float*)d_in[7];
    const float* Wd_s   = (const float*)d_in[8];

    float* y          = (float*)d_out;
    float* out_logits = y + (size_t)T_TOK * H_DIM;
    int*   out_idx    = (int*)(out_logits + (size_t)T_TOK * NEXP);

    char*   ws       = (char*)d_ws;
    int*    counts   = (int*)ws;
    int*    offs     = (int*)(ws + 128);
    int*    tok_list = (int*)(ws + WS_TOK_OFF);
    float*  wgt_list = (float*)(ws + WS_WGT_OFF);
    __bf16* zws      = (__bf16*)(ws + WS_Z_OFF);

    init_counts_kernel<<<1, 32, 0, stream>>>(counts);
    gate_route_kernel<<<T_TOK / 8, 256, 0, stream>>>(
        x, gate_w, bias, out_logits, out_idx, counts, tok_list, wgt_list);
    prefix_kernel<<<1, 1, 0, stream>>>(counts, offs);
    expert_gateup_kernel<<<dim3(I_DIM / 128, T_TOK / M_TILE, NEXP + 1), 512, 0, stream>>>(
        x, Wg, Wu, Wg_s, Wu_s, counts, offs, tok_list, zws);
    // shared expert first with plain stores (defines y), then routed atomics
    down_kernel<true><<<dim3(H_DIM / 128, T_TOK / M_TILE, 1), 512, 0, stream>>>(
        Wd_s, counts, offs, tok_list, wgt_list, zws, y);
    down_kernel<false><<<dim3(H_DIM / 128, T_TOK / M_TILE, NEXP), 512, 0, stream>>>(
        Wd, counts, offs, tok_list, wgt_list, zws, y);
}
